// SwinTokenBlock_27462020891240
// MI455X (gfx1250) — compile-verified, matrix operand order checked by reading, not run on hardware
//
#include <hip/hip_runtime.h>
#include <stdint.h>

// ---------------- types ----------------
typedef __attribute__((ext_vector_type(16))) __bf16 v16bf;
typedef __attribute__((ext_vector_type(8)))  float  v8f;
typedef int v4i_ __attribute__((vector_size(16)));
typedef __attribute__((address_space(1))) v4i_* as1_v4i;
typedef __attribute__((address_space(3))) v4i_* as3_v4i;

#define CC     384
#define NHD    12
#define HD     32
#define NTOK   32768     // B*H*W = 8*64*64
#define NB_    2
#define SHIFT_ 4

#if __has_builtin(__builtin_amdgcn_global_load_async_to_lds_b128)
#define HAVE_ASYNC_LDS 1
#else
#define HAVE_ASYNC_LDS 0
#endif

// f32 -> bf16 round-to-nearest-even
__device__ __forceinline__ unsigned short f2bfu(float f) {
  unsigned u = __float_as_uint(f);
  u += 0x7FFFu + ((u >> 16) & 1u);
  return (unsigned short)(u >> 16);
}

// Build a v16bf WMMA fragment from two contiguous 16-byte chunks
// (CDNA5 16-bit A/B per-lane layout: elems 0..7 = K 8h..8h+7,
//  elems 8..15 = K 16+8h..23+8h).
__device__ __forceinline__ v16bf frag16(const unsigned short* lo,
                                        const unsigned short* hi) {
  union { uint4 q[2]; v16bf v; } u;
  u.q[0] = *(const uint4*)lo;
  u.q[1] = *(const uint4*)hi;
  return u.v;
}

#if HAVE_ASYNC_LDS
// 16B-per-lane async global->LDS DMA (no VGPR data path, ASYNCcnt tracked)
__device__ __forceinline__ void async_cp16(unsigned short* lds,
                                           const unsigned short* g) {
  __builtin_amdgcn_global_load_async_to_lds_b128(
      (as1_v4i)(unsigned short*)g, (as3_v4i)lds, 0, 0);
}
#endif

__device__ __forceinline__ void wait_async0() {
#if HAVE_ASYNC_LDS
#if __has_builtin(__builtin_amdgcn_s_wait_asynccnt)
  __builtin_amdgcn_s_wait_asynccnt(0);
#else
  asm volatile("s_wait_asynccnt 0" ::: "memory");
#endif
#endif
}

__device__ __forceinline__ float blockSum(float v, float* sh) {
  #pragma unroll
  for (int o = 16; o > 0; o >>= 1) v += __shfl_xor(v, o, 32);
  int w = threadIdx.x >> 5;
  if ((threadIdx.x & 31) == 0) sh[w] = v;
  __syncthreads();
  v = sh[0] + sh[1] + sh[2] + sh[3];
  __syncthreads();
  return v;
}

// ---------------------------------------------------------------------------
// One-time weight prep: Wt[n*K + k] = bf16(W[k*N + n])  (transpose + convert)
// ---------------------------------------------------------------------------
__global__ __launch_bounds__(256) void weight_t_bf16(
    const float* __restrict__ W, unsigned short* __restrict__ Wt, int K, int N) {
  int idx = blockIdx.x * 256 + threadIdx.x;   // coalesced read along n
  int k = idx / N, n = idx - k * N;
  Wt[(size_t)n * K + k] = f2bfu(W[idx]);
}

// ---------------------------------------------------------------------------
// Fused outer-LN + LN1 + cyclic shift + window partition.
// ---------------------------------------------------------------------------
__global__ __launch_bounds__(128) void ln_outer_ln1(
    const float* __restrict__ xv,
    const float* __restrict__ og, const float* __restrict__ ob,
    const float* __restrict__ n1g, const float* __restrict__ n1b,
    float* __restrict__ y, unsigned short* __restrict__ hpart, int shift) {
  __shared__ float sh[4];
  const int tok = blockIdx.x, t = threadIdx.x;
  const float* row = xv + (size_t)tok * CC;
  float v0 = row[t], v1 = row[t + 128], v2 = row[t + 256];
  float mean = blockSum(v0 + v1 + v2, sh) * (1.0f / CC);
  float d0 = v0 - mean, d1 = v1 - mean, d2 = v2 - mean;
  float inv = rsqrtf(blockSum(d0*d0 + d1*d1 + d2*d2, sh) * (1.0f / CC) + 1e-5f);
  float y0 = d0 * inv * og[t]       + ob[t];
  float y1 = d1 * inv * og[t + 128] + ob[t + 128];
  float y2 = d2 * inv * og[t + 256] + ob[t + 256];
  float* yr = y + (size_t)tok * CC;
  yr[t] = y0; yr[t + 128] = y1; yr[t + 256] = y2;

  // second LN (attention input)
  float m2 = blockSum(y0 + y1 + y2, sh) * (1.0f / CC);
  float e0 = y0 - m2, e1 = y1 - m2, e2 = y2 - m2;
  float inv2 = rsqrtf(blockSum(e0*e0 + e1*e1 + e2*e2, sh) * (1.0f / CC) + 1e-5f);
  float h0 = e0 * inv2 * n1g[t]       + n1b[t];
  float h1 = e1 * inv2 * n1g[t + 128] + n1b[t + 128];
  float h2 = e2 * inv2 * n1g[t + 256] + n1b[t + 256];

  // roll(-shift) + window partition index
  int b = tok >> 12, p = tok & 4095, oy = p >> 6, ox = p & 63;
  int sy = (oy - shift + 64) & 63, sx = (ox - shift + 64) & 63;
  int prow = ((b << 6) + ((sy >> 3) << 3) + (sx >> 3)) * 64 + ((sy & 7) << 3) + (sx & 7);
  unsigned short* hr = hpart + (size_t)prow * CC;
  hr[t] = f2bfu(h0); hr[t + 128] = f2bfu(h1); hr[t + 256] = f2bfu(h2);
}

// LN2: f32 in -> bf16 out (token order)
__global__ __launch_bounds__(128) void ln2_bf16(
    const float* __restrict__ in, const float* __restrict__ g,
    const float* __restrict__ bb, unsigned short* __restrict__ out) {
  __shared__ float sh[4];
  const int tok = blockIdx.x, t = threadIdx.x;
  const float* row = in + (size_t)tok * CC;
  float v0 = row[t], v1 = row[t + 128], v2 = row[t + 256];
  float mean = blockSum(v0 + v1 + v2, sh) * (1.0f / CC);
  float d0 = v0 - mean, d1 = v1 - mean, d2 = v2 - mean;
  float inv = rsqrtf(blockSum(d0*d0 + d1*d1 + d2*d2, sh) * (1.0f / CC) + 1e-5f);
  unsigned short* o = out + (size_t)tok * CC;
  o[t]       = f2bfu(d0 * inv * g[t]       + bb[t]);
  o[t + 128] = f2bfu(d1 * inv * g[t + 128] + bb[t + 128]);
  o[t + 256] = f2bfu(d2 * inv * g[t + 256] + bb[t + 256]);
}

// ---------------------------------------------------------------------------
// WMMA GEMM: A (MxK, bf16 row-major) x Bt (NxK, bf16 row-major, pre-
// transposed weights) + bias. Block tile 128x128, 8 waves (4x2), wave tile
// 32x64, K step 64 -> 16 WMMA per wave per barrier. Double-buffered LDS
// filled by GLOBAL_LOAD_ASYNC_TO_LDS_B128 (ASYNCcnt) -- no VGPR staging,
// no spills; fallback to register staging if builtin unavailable.
// mode 0: Obf = bf16(act(acc+bias))      (act=1 -> tanh-GELU)
// mode 1: C1 += acc+bias                 (proj residual accumulate)
// mode 2: C2w = C2r + C1 + acc + bias    (double residual, MLP2)
// ---------------------------------------------------------------------------
__global__ __launch_bounds__(256) void gemm_bf16_wmma(
    const unsigned short* __restrict__ A, const unsigned short* __restrict__ Bt,
    const float* __restrict__ bias,
    unsigned short* __restrict__ Obf,
    float* __restrict__ C1, const float* __restrict__ C2r,
    float* __restrict__ C2w,
    int M, int N, int K, int mode, int act) {
  __shared__ __align__(16) unsigned short sA[2][128 * 64];  // [row][k]
  __shared__ __align__(16) unsigned short sB[2][128 * 64];  // [n][k]
  const int tid = threadIdx.x;
  const int wave = tid >> 5, lane = tid & 31;
  const int half = lane >> 4, r = lane & 15;
  const int wm = wave >> 1, wn = wave & 1;            // 4x2 wave grid
  const int row0 = blockIdx.y * 128, col0 = blockIdx.x * 128;

  const v8f zf = {};
  v8f acc[2][4];
  #pragma unroll
  for (int mi = 0; mi < 2; mi++)
    #pragma unroll
    for (int ni = 0; ni < 4; ni++) acc[mi][ni] = zf;

  // staging map: 128 rows x 64 k bf16 = 1024 x 16B chunks; 4 per thread.
  const int srow = tid >> 3;       // base row (step +32 per chunk index)
  const int scb  = tid & 7;        // 8-elem (16B) chunk within row

  // prologue: stage tile 0
#if HAVE_ASYNC_LDS
  #pragma unroll
  for (int i = 0; i < 4; i++) {
    async_cp16(sA[0] + (srow + i * 32) * 64 + scb * 8,
               A  + (size_t)(row0 + srow + i * 32) * K + scb * 8);
    async_cp16(sB[0] + (srow + i * 32) * 64 + scb * 8,
               Bt + (size_t)(col0 + srow + i * 32) * K + scb * 8);
  }
  wait_async0();
#else
  {
    uint4 ra[4], rb[4];
    #pragma unroll
    for (int i = 0; i < 4; i++) {
      ra[i] = *(const uint4*)(A  + (size_t)(row0 + srow + i * 32) * K + scb * 8);
      rb[i] = *(const uint4*)(Bt + (size_t)(col0 + srow + i * 32) * K + scb * 8);
    }
    #pragma unroll
    for (int i = 0; i < 4; i++) {
      *(uint4*)(sA[0] + (srow + i * 32) * 64 + scb * 8) = ra[i];
      *(uint4*)(sB[0] + (srow + i * 32) * 64 + scb * 8) = rb[i];
    }
  }
#endif
  __syncthreads();

  const int nsteps = K >> 6;
  int buf = 0;
  for (int s = 0; s < nsteps; s++) {
    const bool more = (s + 1) < nsteps;
#if HAVE_ASYNC_LDS
    if (more) {
      const int k0 = (s + 1) << 6;
      const int nb = buf ^ 1;
      #pragma unroll
      for (int i = 0; i < 4; i++) {
        async_cp16(sA[nb] + (srow + i * 32) * 64 + scb * 8,
                   A  + (size_t)(row0 + srow + i * 32) * K + k0 + scb * 8);
        async_cp16(sB[nb] + (srow + i * 32) * 64 + scb * 8,
                   Bt + (size_t)(col0 + srow + i * 32) * K + k0 + scb * 8);
      }
    }
#else
    uint4 ra[4], rb[4];
    if (more) {
      const int k0 = (s + 1) << 6;
      #pragma unroll
      for (int i = 0; i < 4; i++) {
        ra[i] = *(const uint4*)(A  + (size_t)(row0 + srow + i * 32) * K + k0 + scb * 8);
        rb[i] = *(const uint4*)(Bt + (size_t)(col0 + srow + i * 32) * K + k0 + scb * 8);
      }
    }
#endif

    // 2 K-subchunks x (2 A-frags reused over 4 B-frags) = 16 WMMA / wave
    #pragma unroll
    for (int kc = 0; kc < 2; kc++) {
      v16bf af[2];
      #pragma unroll
      for (int mi = 0; mi < 2; mi++) {
        int rr = wm * 32 + mi * 16 + r;
        af[mi] = frag16(sA[buf] + rr * 64 + kc * 32 + 8 * half,
                        sA[buf] + rr * 64 + kc * 32 + 16 + 8 * half);
      }
      #pragma unroll
      for (int ni = 0; ni < 4; ni++) {
        int nn = wn * 64 + ni * 16 + r;
        v16bf bfr = frag16(sB[buf] + nn * 64 + kc * 32 + 8 * half,
                           sB[buf] + nn * 64 + kc * 32 + 16 + 8 * half);
        #pragma unroll
        for (int mi = 0; mi < 2; mi++)
          acc[mi][ni] = __builtin_amdgcn_wmma_f32_16x16x32_bf16(
              false, af[mi], false, bfr, (short)0, acc[mi][ni], false, false);
      }
    }

#if HAVE_ASYNC_LDS
    wait_async0();
#else
    if (more) {
      const int nb = buf ^ 1;
      #pragma unroll
      for (int i = 0; i < 4; i++) {
        *(uint4*)(sA[nb] + (srow + i * 32) * 64 + scb * 8) = ra[i];
        *(uint4*)(sB[nb] + (srow + i * 32) * 64 + scb * 8) = rb[i];
      }
    }
#endif
    __syncthreads();
    buf ^= 1;
  }

  // epilogue (C layout: VGPR v, lane half/r -> row v+8*half, col r)
  #pragma unroll
  for (int mi = 0; mi < 2; mi++)
    #pragma unroll
    for (int ni = 0; ni < 4; ni++) {
      int gcol = col0 + wn * 64 + ni * 16 + r;
      float bv = bias[gcol];
      #pragma unroll
      for (int v = 0; v < 8; v++) {
        int grow = row0 + wm * 32 + mi * 16 + v + 8 * half;
        size_t idx = (size_t)grow * N + gcol;
        float val = acc[mi][ni][v] + bv;
        if (mode == 0) {
          if (act) {
            float xx = val;
            float th = tanhf(0.7978845608028654f * (xx + 0.044715f * xx * xx * xx));
            val = 0.5f * xx * (1.0f + th);
          }
          Obf[idx] = f2bfu(val);
        } else if (mode == 1) {
          C1[idx] += val;
        } else {
          C2w[idx] = C2r[idx] + C1[idx] + val;
        }
      }
    }
}

// ---------------------------------------------------------------------------
// Windowed attention: one block per (window, head). 4 waves, each owns a
// 16-row score slab. QK^T: WMMA K=32 with fragments straight from global.
// Softmax: 16-lane shfl_xor reductions. PV: WMMA vs LDS-transposed V.
// Output written to un-shifted token order (reverse+roll fused).
// ---------------------------------------------------------------------------
__global__ __launch_bounds__(128) void swin_attn(
    const unsigned short* __restrict__ qkv, const float* __restrict__ rpb,
    unsigned short* __restrict__ outb, int shift) {
  const int head = blockIdx.x % NHD;
  const int widx = blockIdx.x / NHD;            // 0..511
  const int b = widx >> 6, wl = widx & 63;
  const int wy = wl >> 3, wx = wl & 7;
  const int tid = threadIdx.x;
  const int wave = tid >> 5, lane = tid & 31;
  const int half = lane >> 4, r = lane & 15;

  __shared__ __align__(16) unsigned short sVt[32 * 64];      // [d][token]
  __shared__ __align__(16) unsigned short sP[4 * 16 * 64];   // per-wave P

  const size_t base = (size_t)widx * 64 * 1152;

  // stage V^T with vectorized reads (64 tok x 4 x 8-elem chunks)
  for (int c = tid; c < 256; c += 128) {
    int tok = c >> 2, dblk = (c & 3) * 8;
    uint4 vv = *(const uint4*)(qkv + base + (size_t)tok * 1152 + 768 + head * HD + dblk);
    const unsigned short* pv = (const unsigned short*)&vv;
    #pragma unroll
    for (int j = 0; j < 8; j++) sVt[(dblk + j) * 64 + tok] = pv[j];
  }
  __syncthreads();

  const int m0 = wave * 16;
  const unsigned short* qbase = qkv + base + head * HD;
  const unsigned short* kbase = qkv + base + 384 + head * HD;

  // scores: S(16x64) = q(16x32) @ k^T(32x64)
  v16bf aq = frag16(qbase + (size_t)(m0 + r) * 1152 + 8 * half,
                    qbase + (size_t)(m0 + r) * 1152 + 16 + 8 * half);
  const v8f zf = {};
  v8f sc[4];
  #pragma unroll
  for (int nt = 0; nt < 4; nt++) {
    int n = nt * 16 + r;
    v16bf bk = frag16(kbase + (size_t)n * 1152 + 8 * half,
                      kbase + (size_t)n * 1152 + 16 + 8 * half);
    sc[nt] = __builtin_amdgcn_wmma_f32_16x16x32_bf16(
        false, aq, false, bk, (short)0, zf, false, false);
  }

  // scale + relative-position bias + shift mask
  const float scale = 0.17677669529663687f;  // 1/sqrt(32)
  #pragma unroll
  for (int nt = 0; nt < 4; nt++)
    #pragma unroll
    for (int v = 0; v < 8; v++) {
      int m = m0 + v + 8 * half;
      int n = nt * 16 + r;
      int ty1 = m >> 3, tx1 = m & 7, ty2 = n >> 3, tx2 = n & 7;
      int rel = (ty1 - ty2 + 7) * 15 + (tx1 - tx2 + 7);
      float s = sc[nt][v] * scale + rpb[rel * NHD + head];
      if (shift) {
        int gy1 = wy * 8 + ty1, gx1 = wx * 8 + tx1;
        int gy2 = wy * 8 + ty2, gx2 = wx * 8 + tx2;
        int r1 = (gy1 < 56 ? 0 : (gy1 < 60 ? 1 : 2)) * 3 +
                 (gx1 < 56 ? 0 : (gx1 < 60 ? 1 : 2));
        int r2 = (gy2 < 56 ? 0 : (gy2 < 60 ? 1 : 2)) * 3 +
                 (gx2 < 56 ? 0 : (gx2 < 60 ? 1 : 2));
        if (r1 != r2) s -= 100.0f;
      }
      sc[nt][v] = s;
    }

  // softmax per row (row spread across 16 lanes of this half, 4 accums)
  #pragma unroll
  for (int v = 0; v < 8; v++) {
    float mx = fmaxf(fmaxf(sc[0][v], sc[1][v]), fmaxf(sc[2][v], sc[3][v]));
    #pragma unroll
    for (int o = 8; o > 0; o >>= 1) mx = fmaxf(mx, __shfl_xor(mx, o, 32));
    float sm = 0.0f;
    #pragma unroll
    for (int nt = 0; nt < 4; nt++) {
      float e = __expf(sc[nt][v] - mx);
      sc[nt][v] = e; sm += e;
    }
    #pragma unroll
    for (int o = 8; o > 0; o >>= 1) sm += __shfl_xor(sm, o, 32);
    float invs = 1.0f / sm;
    #pragma unroll
    for (int nt = 0; nt < 4; nt++) sc[nt][v] *= invs;
  }

  // P -> LDS as bf16 (row-major 16x64, wave-private slab)
  unsigned short* Pw = sP + wave * 1024;
  #pragma unroll
  for (int nt = 0; nt < 4; nt++)
    #pragma unroll
    for (int v = 0; v < 8; v++)
      Pw[(v + 8 * half) * 64 + nt * 16 + r] = f2bfu(sc[nt][v]);
  __syncthreads();

  // O(16x32) = P(16x64) @ V(64x32): 2 K-chunks x 2 N-tiles
  v8f o_[2]; o_[0] = zf; o_[1] = zf;
  #pragma unroll
  for (int kc = 0; kc < 2; kc++) {
    v16bf ap = frag16(Pw + r * 64 + kc * 32 + 8 * half,
                      Pw + r * 64 + kc * 32 + 16 + 8 * half);
    #pragma unroll
    for (int nt = 0; nt < 2; nt++) {
      int n = nt * 16 + r;
      v16bf bv = frag16(sVt + n * 64 + kc * 32 + 8 * half,
                        sVt + n * 64 + kc * 32 + 16 + 8 * half);
      o_[nt] = __builtin_amdgcn_wmma_f32_16x16x32_bf16(
          false, ap, false, bv, (short)0, o_[nt], false, false);
    }
  }

  // write to un-shifted token order: reverse-partition + roll(+shift)
  #pragma unroll
  for (int nt = 0; nt < 2; nt++)
    #pragma unroll
    for (int v = 0; v < 8; v++) {
      int m = m0 + v + 8 * half;
      int ty = m >> 3, tx = m & 7;
      int oy = (wy * 8 + ty + shift) & 63;
      int ox = (wx * 8 + tx + shift) & 63;
      size_t idx = ((size_t)b * 4096 + oy * 64 + ox) * CC + head * HD + nt * 16 + r;
      outb[idx] = f2bfu(o_[nt][v]);
    }
}

// ---------------------------------------------------------------------------
extern "C" void kernel_launch(void* const* d_in, const int* in_sizes, int n_in,
                              void* d_out, int out_size, void* d_ws, size_t ws_size,
                              hipStream_t stream) {
  (void)in_sizes; (void)n_in; (void)out_size; (void)ws_size;
  const float* x       = (const float*)d_in[0];
  const float* outer_g = (const float*)d_in[1];
  const float* outer_b = (const float*)d_in[2];
  const float* n1_g    = (const float*)d_in[3];
  const float* n1_b    = (const float*)d_in[4];
  const float* n2_g    = (const float*)d_in[5];
  const float* n2_b    = (const float*)d_in[6];
  const float* qkv_w   = (const float*)d_in[7];
  const float* qkv_b   = (const float*)d_in[8];
  const float* proj_w  = (const float*)d_in[9];
  const float* proj_b  = (const float*)d_in[10];
  const float* mlp_w1  = (const float*)d_in[11];
  const float* mlp_b1  = (const float*)d_in[12];
  const float* mlp_w2  = (const float*)d_in[13];
  const float* mlp_b2  = (const float*)d_in[14];
  const float* rpb     = (const float*)d_in[15];

  char* p = (char*)d_ws;
  float* xv = (float*)p;            p += (size_t)NTOK * CC * 4;   // 48 MB
  float* y  = (float*)p;            p += (size_t)NTOK * CC * 4;   // 48 MB
  unsigned short* hbuf  = (unsigned short*)p; p += (size_t)NTOK * CC * 2;    // 24 MB (h / h2)
  unsigned short* qh    = (unsigned short*)p; p += (size_t)NTOK * 1536 * 2;  // 96 MB (qkv / hidden)
  unsigned short* attnb = (unsigned short*)p; p += (size_t)NTOK * CC * 2;    // 24 MB
  // pre-transposed bf16 weights, per block
  unsigned short* qwT[NB_], *pwT[NB_], *w1T[NB_], *w2T[NB_];
  for (int i = 0; i < NB_; i++) {
    qwT[i] = (unsigned short*)p; p += (size_t)1152 * CC * 2;
    pwT[i] = (unsigned short*)p; p += (size_t)CC * CC * 2;
    w1T[i] = (unsigned short*)p; p += (size_t)1536 * CC * 2;
    w2T[i] = (unsigned short*)p; p += (size_t)CC * 1536 * 2;
  }

  // one-time weight transpose+convert (L2-resident afterwards)
  for (int i = 0; i < NB_; i++) {
    weight_t_bf16<<<(CC * 1152) / 256, 256, 0, stream>>>(
        qkv_w + (size_t)i * CC * 1152, qwT[i], CC, 1152);
    weight_t_bf16<<<(CC * CC) / 256, 256, 0, stream>>>(
        proj_w + (size_t)i * CC * CC, pwT[i], CC, CC);
    weight_t_bf16<<<(CC * 1536) / 256, 256, 0, stream>>>(
        mlp_w1 + (size_t)i * CC * 1536, w1T[i], CC, 1536);
    weight_t_bf16<<<(1536 * CC) / 256, 256, 0, stream>>>(
        mlp_w2 + (size_t)i * 1536 * CC, w2T[i], 1536, CC);
  }

  (void)hipMemcpyAsync(xv, x, (size_t)NTOK * CC * 4, hipMemcpyDeviceToDevice, stream);

  const dim3 gQKV(1152 / 128, NTOK / 128);
  const dim3 gPROJ(CC / 128, NTOK / 128);
  const dim3 gMLP1(1536 / 128, NTOK / 128);

  for (int i = 0; i < NB_; i++) {
    const int shift = (i & 1) ? SHIFT_ : 0;
    const float* og  = outer_g + i * CC;
    const float* obv = outer_b + i * CC;
    const float* g1  = n1_g + i * CC;
    const float* b1n = n1_b + i * CC;
    const float* g2  = n2_g + i * CC;
    const float* b2n = n2_b + i * CC;
    const float* qb  = qkv_b + i * 1152;
    const float* pb  = proj_b + i * CC;
    const float* bb1 = mlp_b1 + i * 1536;
    const float* bb2 = mlp_b2 + i * CC;
    const float* rp  = rpb + (size_t)i * 225 * NHD;

    ln_outer_ln1<<<NTOK, 128, 0, stream>>>(xv, og, obv, g1, b1n, y, hbuf, shift);

    gemm_bf16_wmma<<<gQKV, 256, 0, stream>>>(hbuf, qwT[i], qb, qh,
        nullptr, nullptr, nullptr, NTOK, 1152, CC, 0, 0);

    swin_attn<<<512 * NHD, 128, 0, stream>>>(qh, rp, attnb, shift);

    gemm_bf16_wmma<<<gPROJ, 256, 0, stream>>>(attnb, pwT[i], pb, nullptr,
        y, nullptr, nullptr, NTOK, CC, CC, 1, 0);

    ln2_bf16<<<NTOK, 128, 0, stream>>>(y, g2, b2n, hbuf);

    gemm_bf16_wmma<<<gMLP1, 256, 0, stream>>>(hbuf, w1T[i], bb1, qh,
        nullptr, nullptr, nullptr, NTOK, 1536, CC, 0, 1);

    float* cw = (i == NB_ - 1) ? (float*)d_out : xv;
    gemm_bf16_wmma<<<gPROJ, 256, 0, stream>>>(qh, w2T[i], bb2, nullptr,
        y, xv, cw, NTOK, CC, 1536, 2, 0);
  }
}
